// BrainGAT_54408645706224
// MI455X (gfx1250) — matrix operand order repaired and checked by reading, not compiled, not run on hardware
//
#include <hip/hip_runtime.h>
#include <hip/hip_bf16.h>
#include <hip/hip_fp16.h>

#define N_NODES  50000
#define NPAD     50048           // multiple of 64
#define N_EDGES  800000
#define EA       (N_EDGES + N_NODES)   // edges incl. self loops
#define N_GRAPHS 256
#define SLOPE    0.2f
#define BN_EPS   1e-5f

typedef __attribute__((ext_vector_type(16))) _Float16 v16h;
typedef __attribute__((ext_vector_type(8)))  float    v8f;

// ---- monotonic float <-> uint encoding for atomicMax on floats ----
__device__ __forceinline__ unsigned fenc(float f) {
  unsigned u = __float_as_uint(f);
  return (u & 0x80000000u) ? ~u : (u | 0x80000000u);
}
__device__ __forceinline__ float fdec(unsigned u) {
  return __uint_as_float((u & 0x80000000u) ? (u & 0x7FFFFFFFu) : ~u);
}

// =====================================================================
// WMMA GEMM:  H[M,N] = X[M,256] * W[256,N]  (f32 in, f16 WMMA, f32 accum)
// Block = 256 threads = 8 waves; block tile 64(M) x 32(N); K fully unrolled.
// B tile staged in LDS pre-swizzled into WMMA fragment order so each lane
// reads its 16 halves as one aligned 32-byte chunk (2 x ds_load_b128).
// A rows loaded unconditionally from a clamped pointer (no predication);
// out-of-range rows are discarded at writeback.
// =====================================================================
__global__ __launch_bounds__(256)
void k_gemm_wmma(const float* __restrict__ X, const float* __restrict__ W,
                 float* __restrict__ H, int M, int N) {
  constexpr int K = 256;
  __shared__ alignas(32) _Float16 Wlds[K * 32];   // 16 KB, fragment order

  const int tid = threadIdx.x;
  const int nBase = blockIdx.y * 32;

  // Stage B tile in fragment order:
  //   element (k, c):  kb = k/32, rem = k%32, hs = (rem/8)%2, j = (rem/16)*8 + rem%8
  //   dst = (((kb*2 + hs)*32 + c)*16 + j)
  for (int i = tid; i < K * 32; i += 256) {
    int k = i >> 5, c = i & 31;
    int kb = k >> 5, rem = k & 31, hs = (rem >> 3) & 1, j = ((rem >> 4) << 3) | (rem & 7);
    Wlds[((((kb << 1) + hs) * 32 + c) << 4) + j] = (_Float16)W[k * N + nBase + c];
  }
  __syncthreads();

  const int wave = tid >> 5;
  const int lane = tid & 31;
  const int mt = wave & 3;            // 4 tiles along M
  const int nt = wave >> 2;           // 2 tiles along N
  const int mBase = blockIdx.x * 64 + mt * 16;
  const int nloc  = nt * 16 + (lane & 15);

  const int m = mBase + (lane & 15);
  const float* rowA = X + (size_t)(m < M ? m : 0) * K;  // clamped, branch-free
  const int kbA = (lane < 16) ? 0 : 8;                  // ISA 16-bit A layout
  const int hsB = (lane < 16) ? 0 : 1;                  // B K-half per lane

  v8f acc = {};
#pragma unroll
  for (int kb = 0; kb < K / 32; ++kb) {
    const int k0 = kb * 32;
    const float4 p0 = *(const float4*)(rowA + k0 + kbA);
    const float4 p1 = *(const float4*)(rowA + k0 + kbA + 4);
    const float4 p2 = *(const float4*)(rowA + k0 + 16 + kbA);
    const float4 p3 = *(const float4*)(rowA + k0 + 16 + kbA + 4);
    v16h a;
    a[0]  = (_Float16)p0.x; a[1]  = (_Float16)p0.y;
    a[2]  = (_Float16)p0.z; a[3]  = (_Float16)p0.w;
    a[4]  = (_Float16)p1.x; a[5]  = (_Float16)p1.y;
    a[6]  = (_Float16)p1.z; a[7]  = (_Float16)p1.w;
    a[8]  = (_Float16)p2.x; a[9]  = (_Float16)p2.y;
    a[10] = (_Float16)p2.z; a[11] = (_Float16)p2.w;
    a[12] = (_Float16)p3.x; a[13] = (_Float16)p3.y;
    a[14] = (_Float16)p3.z; a[15] = (_Float16)p3.w;

    const v16h b = *(const v16h*)&Wlds[((((kb << 1) + hsB) * 32 + nloc) << 4)];

    acc = __builtin_amdgcn_wmma_f32_16x16x32_f16(false, a, false, b,
                                                 (short)0, acc, false, false);
  }

  const int nG = nBase + nloc;
  const int rsel = (lane < 16) ? 0 : 8;
#pragma unroll
  for (int r = 0; r < 8; ++r) {
    int mr = mBase + rsel + r;
    if (mr < M) H[(size_t)mr * N + nG] = acc[r];
  }
}

// =====================================================================
// Per-layer init: zero aggregation buffer, reset softmax max/den
// =====================================================================
__global__ void k_init_layer(float* __restrict__ G, unsigned* __restrict__ emax,
                             float* __restrict__ den, int nG, int nH) {
  int idx = blockIdx.x * blockDim.x + threadIdx.x;
  if (idx < nG) G[idx] = 0.f;
  if (idx < nH) { emax[idx] = fenc(-3.0e38f); den[idx] = 0.f; }
}

// =====================================================================
// Attention coefficients: a_src[n,h] = <H[n,h,:], att_src[h,:]> (C=64)
// =====================================================================
__global__ void k_att(const float* __restrict__ H, const float* __restrict__ attS,
                      const float* __restrict__ attD, float* __restrict__ aS,
                      float* __restrict__ aD, int heads) {
  int idx = blockIdx.x * blockDim.x + threadIdx.x;
  if (idx >= N_NODES * heads) return;
  int n = idx / heads, h = idx - n * heads;
  const int F = heads * 64;
  const float* row = H + (size_t)n * F + h * 64;
  const float* ws = attS + h * 64;
  const float* wd = attD + h * 64;
  float s = 0.f, d = 0.f;
#pragma unroll 8
  for (int c = 0; c < 64; ++c) { float v = row[c]; s += v * ws[c]; d += v * wd[c]; }
  aS[idx] = s;
  aD[idx] = d;
}

__device__ __forceinline__ void edge_nodes(const int* __restrict__ ei, int e,
                                           int& s, int& d) {
  if (e < N_EDGES) { s = ei[e]; d = ei[N_EDGES + e]; }
  else             { s = e - N_EDGES; d = s; }            // self loop
}

// edge softmax pass 1: segment max over dst
__global__ void k_edge_max(const int* __restrict__ ei, const float* __restrict__ aS,
                           const float* __restrict__ aD, unsigned* __restrict__ emax,
                           int heads) {
  int idx = blockIdx.x * blockDim.x + threadIdx.x;
  if (idx >= EA * heads) return;
  int e = idx / heads, h = idx - e * heads;
  int s, d; edge_nodes(ei, e, s, d);
  float v = aS[s * heads + h] + aD[d * heads + h];
  v = v > 0.f ? v : SLOPE * v;                            // leaky relu
  atomicMax(&emax[d * heads + h], fenc(v));
}

// edge softmax pass 2: exp(e - max), segment sum of denominators
__global__ void k_edge_exp(const int* __restrict__ ei, const float* __restrict__ aS,
                           const float* __restrict__ aD, const unsigned* __restrict__ emax,
                           float* __restrict__ ee, float* __restrict__ den, int heads) {
  int idx = blockIdx.x * blockDim.x + threadIdx.x;
  if (idx >= EA * heads) return;
  int e = idx / heads, h = idx - e * heads;
  int s, d; edge_nodes(ei, e, s, d);
  float v = aS[s * heads + h] + aD[d * heads + h];
  v = v > 0.f ? v : SLOPE * v;
  float x = __expf(v - fdec(emax[d * heads + h]));
  ee[idx] = x;
  atomicAdd(&den[d * heads + h], x);
}

// edge softmax pass 3: out[dst] += alpha * h[src]  (thread per edge*channel)
__global__ void k_edge_agg(const int* __restrict__ ei, const float* __restrict__ H,
                           const float* __restrict__ ee, const float* __restrict__ den,
                           float* __restrict__ G, int heads) {
  const int F = heads * 64;
  int idx = blockIdx.x * blockDim.x + threadIdx.x;
  if (idx >= EA * F) return;
  int e = idx / F, c = idx - e * F;
  int s, d; edge_nodes(ei, e, s, d);
  int h = c >> 6;
  const float* hrow = H + (size_t)s * F;
  __builtin_prefetch(hrow + c, 0, 0);                     // global_prefetch_b8
  float alpha = ee[e * heads + h] / den[d * heads + h];
  atomicAdd(&G[(size_t)d * F + c], alpha * hrow[c]);
}

// fused bias + batch-norm (eval) + ELU, in place on G
__global__ void k_bias_bn_elu(float* __restrict__ G, const float* __restrict__ bias,
                              const float* __restrict__ gam, const float* __restrict__ bet,
                              const float* __restrict__ mu, const float* __restrict__ var,
                              int F) {
  int idx = blockIdx.x * blockDim.x + threadIdx.x;
  if (idx >= N_NODES * F) return;
  int f = idx % F;
  float v = G[idx] + bias[f];
  v = (v - mu[f]) * rsqrtf(var[f] + BN_EPS) * gam[f] + bet[f];
  G[idx] = v > 0.f ? v : (__expf(v) - 1.f);               // ELU
}

// =====================================================================
// Graph pooling (mean + max over batch segments) and MLP head
// =====================================================================
__global__ void k_pool_init(float* __restrict__ psum, unsigned* __restrict__ pmax,
                            float* __restrict__ pcnt) {
  int idx = blockIdx.x * blockDim.x + threadIdx.x;
  if (idx < N_GRAPHS * 64) { psum[idx] = 0.f; pmax[idx] = fenc(-3.0e38f); }
  if (idx < N_GRAPHS) pcnt[idx] = 0.f;
}

__global__ void k_pool(const float* __restrict__ X, const int* __restrict__ batch,
                       float* __restrict__ psum, unsigned* __restrict__ pmax,
                       float* __restrict__ pcnt) {
  int idx = blockIdx.x * blockDim.x + threadIdx.x;
  if (idx >= N_NODES * 64) return;
  int n = idx >> 6, f = idx & 63;
  int b = batch[n];
  float v = X[idx];
  atomicAdd(&psum[b * 64 + f], v);
  atomicMax(&pmax[b * 64 + f], fenc(v));
  if (f == 0) atomicAdd(&pcnt[b], 1.0f);
}

__global__ __launch_bounds__(128)
void k_head(const float* __restrict__ psum, const unsigned* __restrict__ pmax,
            const float* __restrict__ pcnt,
            const float* __restrict__ W1, const float* __restrict__ b1,
            const float* __restrict__ W2, const float* __restrict__ b2,
            float* __restrict__ out) {
  __shared__ float g[128];
  __shared__ float hid[64];
  const int b = blockIdx.x;
  const int t = threadIdx.x;
  const float cnt = pcnt[b];
  {
    float v;
    if (t < 64) v = psum[b * 64 + t] / fmaxf(cnt, 1.0f);
    else {
      float mx = fdec(pmax[b * 64 + (t - 64)]);
      v = (cnt > 0.f) ? mx : 0.f;
    }
    g[t] = v;
  }
  __syncthreads();
  if (t < 64) {
    float acc = b1[t];
#pragma unroll 4
    for (int i = 0; i < 128; ++i) acc += g[i] * W1[i * 64 + t];
    hid[t] = fmaxf(acc, 0.f);                             // ReLU
  }
  __syncthreads();
  if (t < 2) {
    float acc = b2[t];
#pragma unroll 8
    for (int j = 0; j < 64; ++j) acc += hid[j] * W2[j * 2 + t];
    out[b * 2 + t] = acc;
  }
}

// =====================================================================
// Host side
// =====================================================================
static void run_gat_layer(const float* Xin, const int* ei,
                          const float* W, const float* attS, const float* attD,
                          const float* bias, const float* gam, const float* bet,
                          const float* mu, const float* var, int heads,
                          float* H, float* G, float* aS, float* aD,
                          unsigned* emax, float* den, float* ee,
                          hipStream_t stream) {
  const int Fo = heads * 64;
  dim3 gg(NPAD / 64, Fo / 32);
  k_gemm_wmma<<<gg, 256, 0, stream>>>(Xin, W, H, N_NODES, Fo);

  const int nG = NPAD * Fo, nH = NPAD * heads;
  k_init_layer<<<(nG + 255) / 256, 256, 0, stream>>>(G, emax, den, nG, nH);

  k_att<<<(N_NODES * heads + 255) / 256, 256, 0, stream>>>(H, attS, attD, aS, aD, heads);
  k_edge_max<<<(EA * heads + 255) / 256, 256, 0, stream>>>(ei, aS, aD, emax, heads);
  k_edge_exp<<<(EA * heads + 255) / 256, 256, 0, stream>>>(ei, aS, aD, emax, ee, den, heads);
  k_edge_agg<<<(unsigned)(((long long)EA * Fo + 255) / 256), 256, 0, stream>>>(ei, H, ee, den, G, heads);
  k_bias_bn_elu<<<(N_NODES * Fo + 255) / 256, 256, 0, stream>>>(G, bias, gam, bet, mu, var, Fo);
}

extern "C" void kernel_launch(void* const* d_in, const int* in_sizes, int n_in,
                              void* d_out, int out_size, void* d_ws, size_t ws_size,
                              hipStream_t stream) {
  // inputs in setup_inputs() dict order, params flattened recursively
  const float* x     = (const float*)d_in[0];
  const int*   ei    = (const int*)d_in[1];
  const int*   batch = (const int*)d_in[2];
  const float* c1W  = (const float*)d_in[3];
  const float* c1aS = (const float*)d_in[4];
  const float* c1aD = (const float*)d_in[5];
  const float* c1b  = (const float*)d_in[6];
  const float* c2W  = (const float*)d_in[7];
  const float* c2aS = (const float*)d_in[8];
  const float* c2aD = (const float*)d_in[9];
  const float* c2b  = (const float*)d_in[10];
  const float* c3W  = (const float*)d_in[11];
  const float* c3aS = (const float*)d_in[12];
  const float* c3aD = (const float*)d_in[13];
  const float* c3b  = (const float*)d_in[14];
  const float* bn1g = (const float*)d_in[15];
  const float* bn1b = (const float*)d_in[16];
  const float* bn1m = (const float*)d_in[17];
  const float* bn1v = (const float*)d_in[18];
  const float* bn2g = (const float*)d_in[19];
  const float* bn2b = (const float*)d_in[20];
  const float* bn2m = (const float*)d_in[21];
  const float* bn2v = (const float*)d_in[22];
  const float* bn3g = (const float*)d_in[23];
  const float* bn3b = (const float*)d_in[24];
  const float* bn3m = (const float*)d_in[25];
  const float* bn3v = (const float*)d_in[26];
  const float* fc1W = (const float*)d_in[27];
  const float* fc1b = (const float*)d_in[28];
  const float* fc2W = (const float*)d_in[29];
  const float* fc2b = (const float*)d_in[30];
  float* out = (float*)d_out;

  // workspace layout (floats)
  float* ws = (float*)d_ws;
  size_t o = 0;
  float*    H    = ws + o; o += (size_t)NPAD * 256;
  float*    G    = ws + o; o += (size_t)NPAD * 256;
  float*    aS   = ws + o; o += (size_t)NPAD * 4;
  float*    aD   = ws + o; o += (size_t)NPAD * 4;
  unsigned* emax = (unsigned*)(ws + o); o += (size_t)NPAD * 4;
  float*    den  = ws + o; o += (size_t)NPAD * 4;
  float*    ee   = ws + o; o += (size_t)EA * 4;
  float*    psum = ws + o; o += (size_t)N_GRAPHS * 64;
  unsigned* pmax = (unsigned*)(ws + o); o += (size_t)N_GRAPHS * 64;
  float*    pcnt = ws + o; o += N_GRAPHS;

  // layer 1: x[50000,256] -> G[50000,256]
  run_gat_layer(x, ei, c1W, c1aS, c1aD, c1b, bn1g, bn1b, bn1m, bn1v, 4,
                H, G, aS, aD, emax, den, ee, stream);
  // layer 2: G -> G (GEMM consumes G into H before G is re-initialized)
  run_gat_layer(G, ei, c2W, c2aS, c2aD, c2b, bn2g, bn2b, bn2m, bn2v, 4,
                H, G, aS, aD, emax, den, ee, stream);
  // layer 3: G[.,256] -> G[.,64], heads=1
  run_gat_layer(G, ei, c3W, c3aS, c3aD, c3b, bn3g, bn3b, bn3m, bn3v, 1,
                H, G, aS, aD, emax, den, ee, stream);

  // pooling + MLP head
  k_pool_init<<<(N_GRAPHS * 64 + 255) / 256, 256, 0, stream>>>(psum, pmax, pcnt);
  k_pool<<<(N_NODES * 64 + 255) / 256, 256, 0, stream>>>(G, batch, psum, pmax, pcnt);
  k_head<<<N_GRAPHS, 128, 0, stream>>>(psum, pmax, pcnt, fc1W, fc1b, fc2W, fc2b, out);
}
